// TripletLoss_1477468750076
// MI455X (gfx1250) — matrix operand order repaired and checked by reading, not compiled, not run on hardware
//
#include <hip/hip_runtime.h>

#define MARGIN 0.2f
#define TM   256   // block tile rows (M)
#define TN   128   // block tile cols (N)
#define KB   32
#define SA   40    // LDS row stride in ushorts (80B: 16B-aligned, bank-spread)
#define SB   40

typedef __bf16 bf16_t;
typedef bf16_t v16bf __attribute__((ext_vector_type(16)));
typedef bf16_t v8bf  __attribute__((ext_vector_type(8)));
typedef float  v8f   __attribute__((ext_vector_type(8)));
typedef unsigned short v4u16 __attribute__((ext_vector_type(4)));

// ---- async global->LDS staging path (CDNA5) ----
// Builtin signature (from probe diagnostics): first param is
// 'int __attribute__((vector_size(16))) addrspace(1)*' (prints as "int4 __device__*").
typedef int v4i32 __attribute__((vector_size(16)));
typedef __attribute__((address_space(1))) v4i32 g_v4i32;
typedef __attribute__((address_space(3))) v4i32 l_v4i32;

#if defined(__AMDGCN__) && __has_builtin(__builtin_amdgcn_global_load_async_to_lds_b128)
#define ASYNC_STAGING 1
#else
#define ASYNC_STAGING 0
#endif

#if ASYNC_STAGING
#define NBUF 2
// generic->AS pointers via integer casts (LDS generic ptr low 32 bits = LDS offset)
#define GLD_ASYNC_B128(gp, lp)                                             \
  __builtin_amdgcn_global_load_async_to_lds_b128(                          \
      (g_v4i32*)(uintptr_t)(gp), (l_v4i32*)(unsigned)(uintptr_t)(lp), 0, 0)
#if __has_builtin(__builtin_amdgcn_s_wait_asynccnt)
#define WAIT_ASYNC(n) __builtin_amdgcn_s_wait_asynccnt(n)
#else
#define WAIT_ASYNC(n) asm volatile("s_wait_asynccnt %0" :: "n"(n) : "memory")
#endif
#else
#define NBUF 1
#endif

// fp32 -> bf16 round-to-nearest-even (inputs are finite normals)
__device__ __forceinline__ unsigned short f2bf(float f){
  unsigned int u = __float_as_uint(f);
  u += 0x7FFFu + ((u >> 16) & 1u);
  return (unsigned short)(u >> 16);
}
__device__ __forceinline__ float bf2f(unsigned short h){
  return __uint_as_float(((unsigned int)h) << 16);
}

// float atomic max via monotone int/uint encoding (works with -inf init)
__device__ __forceinline__ void atomicMaxF(float* addr, float v){
  if (v >= 0.0f) atomicMax((int*)addr, __float_as_int(v));
  else           atomicMin((unsigned int*)addr, __float_as_uint(v));
}

__device__ __forceinline__ v16bf cat8(v8bf a, v8bf b){
  return __builtin_shufflevector(a, b, 0,1,2,3,4,5,6,7,8,9,10,11,12,13,14,15);
}

__device__ __forceinline__ v8f wmma_bf16(v16bf a, v16bf b, v8f c){
  // (neg_a, A, neg_b, B, c_mod, C, reuse_a, reuse_b)
  return __builtin_amdgcn_wmma_f32_16x16x32_bf16(false, a, false, b, (short)0, c, false, false);
}

// ---------------- one-time fp32 -> bf16 hi/lo split ----------------
__global__ __launch_bounds__(256) void split_bf16(
    const float* __restrict__ img, const float* __restrict__ txt,
    unsigned short* __restrict__ aHi, unsigned short* __restrict__ aLo,
    unsigned short* __restrict__ bHi, unsigned short* __restrict__ bLo,
    int total4)
{
  int i4 = blockIdx.x * blockDim.x + threadIdx.x;
  if (i4 >= total4) return;
  size_t off = (size_t)i4 * 4;

  float4 fa = *(const float4*)(img + off);
  float4 fb = *(const float4*)(txt + off);
  float va[4] = {fa.x, fa.y, fa.z, fa.w};
  float vb[4] = {fb.x, fb.y, fb.z, fb.w};
  v4u16 ah, al, bh, bl;
  #pragma unroll
  for (int j = 0; j < 4; ++j){
    unsigned short h = f2bf(va[j]);
    ah[j] = h; al[j] = f2bf(va[j] - bf2f(h));
    h = f2bf(vb[j]);
    bh[j] = h; bl[j] = f2bf(vb[j] - bf2f(h));
  }
  *(v4u16*)(aHi + off) = ah;
  *(v4u16*)(aLo + off) = al;
  *(v4u16*)(bHi + off) = bh;
  *(v4u16*)(bLo + off) = bl;
}

// ---------------- main GEMM + masked row/col max ----------------
__global__ __launch_bounds__(256) void tile_gemm_max(
    const unsigned short* __restrict__ aHi, const unsigned short* __restrict__ aLo,
    const unsigned short* __restrict__ bHi, const unsigned short* __restrict__ bLo,
    float* __restrict__ row_max, float* __restrict__ col_max,
    float* __restrict__ pos, int B, int D)
{
  __shared__ unsigned short sAhi[NBUF][TM*SA];
  __shared__ unsigned short sAlo[NBUF][TM*SA];
  __shared__ unsigned short sBhi[NBUF][TN*SB];
  __shared__ unsigned short sBlo[NBUF][TN*SB];

  const int tid  = threadIdx.x;
  const int lane = tid & 31;
  const int wave = tid >> 5;
  const int half = lane >> 4;   // lanes 0-15 vs 16-31
  const int l16  = lane & 15;
  const int wr   = wave >> 1;   // 0..3 : 64 M-rows per wave
  const int wc   = wave & 1;    // 0..1 : 64 N-cols per wave

  const int bm = blockIdx.y * TM;
  const int bn = blockIdx.x * TN;

  // staging: A rows 0..255 -> thread t handles full 64B row (4 x b128) of hi and lo
  //          B rows 0..127 -> thread t handles half row (2 x b128) of hi and lo
  const unsigned short* gAh = aHi + (size_t)(bm + tid) * D;
  const unsigned short* gAl = aLo + (size_t)(bm + tid) * D;
  const int brow = tid >> 1;
  const int bsel = (tid & 1) * 16;          // element offset within the 32-elem row
  const unsigned short* gBh = bHi + (size_t)(bn + brow) * D + bsel;
  const unsigned short* gBl = bLo + (size_t)(bn + brow) * D + bsel;

  v8f acc[4][4] = {};
  const int nk = D / KB;

#if ASYNC_STAGING
  // all data movement global->LDS on the ASYNCcnt engine; no data VGPRs
  auto issue_batch = [&](int bf, int k0){
    #pragma unroll
    for (int c = 0; c < 4; ++c){
      GLD_ASYNC_B128(gAh + k0 + c*8, &sAhi[bf][tid*SA + c*8]);
      GLD_ASYNC_B128(gAl + k0 + c*8, &sAlo[bf][tid*SA + c*8]);
    }
    #pragma unroll
    for (int c = 0; c < 2; ++c){
      GLD_ASYNC_B128(gBh + k0 + c*8, &sBhi[bf][brow*SB + bsel + c*8]);
      GLD_ASYNC_B128(gBl + k0 + c*8, &sBlo[bf][brow*SB + bsel + c*8]);
    }
  };
  issue_batch(0, 0);
#else
  uint4 rAh[4], rAl[4], rBh[2], rBl[2];
  #pragma unroll
  for (int c = 0; c < 4; ++c){
    rAh[c] = *(const uint4*)(gAh + c*8);
    rAl[c] = *(const uint4*)(gAl + c*8);
  }
  #pragma unroll
  for (int c = 0; c < 2; ++c){
    rBh[c] = *(const uint4*)(gBh + c*8);
    rBl[c] = *(const uint4*)(gBl + c*8);
  }
#endif

  for (int kt = 0; kt < nk; ++kt){
#if ASYNC_STAGING
    // issue next batch into the other buffer (final iter: harmless wrap to k=0)
    {
      int nkt = (kt + 1 == nk) ? 0 : kt + 1;
      issue_batch(nkt & 1, nkt * KB);
    }
    WAIT_ASYNC(12);            // 12 newest outstanding => previous batch landed
    __syncthreads();
    const unsigned short* bAhi = sAhi[kt & 1];
    const unsigned short* bAlo = sAlo[kt & 1];
    const unsigned short* bBhi = sBhi[kt & 1];
    const unsigned short* bBlo = sBlo[kt & 1];
#else
    __syncthreads();
    #pragma unroll
    for (int c = 0; c < 4; ++c){
      *(uint4*)&sAhi[0][tid*SA + c*8] = rAh[c];
      *(uint4*)&sAlo[0][tid*SA + c*8] = rAl[c];
    }
    #pragma unroll
    for (int c = 0; c < 2; ++c){
      *(uint4*)&sBhi[0][brow*SB + bsel + c*8] = rBh[c];
      *(uint4*)&sBlo[0][brow*SB + bsel + c*8] = rBl[c];
    }
    __syncthreads();
    if (kt + 1 < nk){
      int k0 = (kt + 1) * KB;
      #pragma unroll
      for (int c = 0; c < 4; ++c){
        rAh[c] = *(const uint4*)(gAh + k0 + c*8);
        rAl[c] = *(const uint4*)(gAl + k0 + c*8);
      }
      #pragma unroll
      for (int c = 0; c < 2; ++c){
        rBh[c] = *(const uint4*)(gBh + k0 + c*8);
        rBl[c] = *(const uint4*)(gBl + k0 + c*8);
      }
    }
    const unsigned short* bAhi = sAhi[0];
    const unsigned short* bAlo = sAlo[0];
    const unsigned short* bBhi = sBhi[0];
    const unsigned short* bBlo = sBlo[0];
#endif

    // B fragments: lane l16 = column; lanes 0-15 hold K 0..15, lanes 16-31 K 16..31
    v16bf fbh[4], fbl[4];
    #pragma unroll
    for (int nf = 0; nf < 4; ++nf){
      int br = wc*64 + nf*16 + l16;
      fbh[nf] = cat8(*(const v8bf*)&bBhi[br*SB + half*16],
                     *(const v8bf*)&bBhi[br*SB + half*16 + 8]);
      fbl[nf] = cat8(*(const v8bf*)&bBlo[br*SB + half*16],
                     *(const v8bf*)&bBlo[br*SB + half*16 + 8]);
    }
    // A fragments: lane l16 = row; lanes 0-15 hold K 0..15, lanes 16-31 K 16..31
    #pragma unroll
    for (int mf = 0; mf < 4; ++mf){
      int ar = wr*64 + mf*16 + l16;
      v16bf ah = cat8(*(const v8bf*)&bAhi[ar*SA + half*16],
                      *(const v8bf*)&bAhi[ar*SA + half*16 + 8]);
      v16bf al = cat8(*(const v8bf*)&bAlo[ar*SA + half*16],
                      *(const v8bf*)&bAlo[ar*SA + half*16 + 8]);
      #pragma unroll
      for (int nf = 0; nf < 4; ++nf){
        acc[mf][nf] = wmma_bf16(ah, fbh[nf], acc[mf][nf]);  // hi*hi
        acc[mf][nf] = wmma_bf16(ah, fbl[nf], acc[mf][nf]);  // hi*lo
        acc[mf][nf] = wmma_bf16(al, fbh[nf], acc[mf][nf]);  // lo*hi
      }
    }
#if ASYNC_STAGING
    __syncthreads();   // compute(kt) done before buf[kt&1] is overwritten at kt+2
#endif
  }

  // --- epilogue: diagonal capture + masked row/col maxima ---
  const float NINF = -__builtin_inff();
  const int gM0 = bm + wr*64;
  const int gN0 = bn + wc*64;

  // C layout: VGPR r, lanes 0-15 -> M=r, lanes 16-31 -> M=r+8; N = l16
  #pragma unroll
  for (int mf = 0; mf < 4; ++mf){
    #pragma unroll
    for (int r = 0; r < 8; ++r){
      int gi = gM0 + mf*16 + half*8 + r;
      float rv = NINF;
      #pragma unroll
      for (int nf = 0; nf < 4; ++nf){
        int gj = gN0 + nf*16 + l16;
        float v = acc[mf][nf][r];
        if (gi == gj){ pos[gi] = v; v = NINF; }
        rv = fmaxf(rv, v);
      }
      rv = fmaxf(rv, __shfl_xor(rv, 1));
      rv = fmaxf(rv, __shfl_xor(rv, 2));
      rv = fmaxf(rv, __shfl_xor(rv, 4));
      rv = fmaxf(rv, __shfl_xor(rv, 8));
      if (l16 == 0) atomicMaxF(&row_max[gi], rv);
    }
  }

  #pragma unroll
  for (int nf = 0; nf < 4; ++nf){
    int gj = gN0 + nf*16 + l16;
    float cv = NINF;
    #pragma unroll
    for (int mf = 0; mf < 4; ++mf){
      #pragma unroll
      for (int r = 0; r < 8; ++r){
        int gi = gM0 + mf*16 + half*8 + r;
        float v = acc[mf][nf][r];
        if (gi == gj) v = NINF;
        cv = fmaxf(cv, v);
      }
    }
    cv = fmaxf(cv, __shfl_xor(cv, 16));
    if (half == 0) atomicMaxF(&col_max[gj], cv);
  }
}

__global__ void init_ws(float* rm, float* cm, float* pos, int B){
  int i = blockIdx.x * blockDim.x + threadIdx.x;
  if (i < B){
    rm[i] = -__builtin_inff();
    cm[i] = -__builtin_inff();
    pos[i] = 0.0f;
  }
}

__global__ __launch_bounds__(256) void finalize_loss(
    const float* __restrict__ rm, const float* __restrict__ cm,
    const float* __restrict__ pos, float* __restrict__ out, int B)
{
  __shared__ float s1s[256], s2s[256];
  float s1 = 0.f, s2 = 0.f;
  for (int i = threadIdx.x; i < B; i += 256){
    float p = pos[i];
    s1 += fmaxf(MARGIN + rm[i] - p, 0.f);
    s2 += fmaxf(MARGIN + cm[i] - p, 0.f);
  }
  s1s[threadIdx.x] = s1; s2s[threadIdx.x] = s2;
  __syncthreads();
  for (int s = 128; s > 0; s >>= 1){
    if ((int)threadIdx.x < s){
      s1s[threadIdx.x] += s1s[threadIdx.x + s];
      s2s[threadIdx.x] += s2s[threadIdx.x + s];
    }
    __syncthreads();
  }
  if (threadIdx.x == 0){
    float l1 = s1s[0] / (float)B;
    float l2 = s2s[0] / (float)B;
    out[0] = l1 + l2;  // total
    out[1] = l1;       // loss_i2t
    out[2] = l2;       // loss_t2i
  }
}

extern "C" void kernel_launch(void* const* d_in, const int* in_sizes, int n_in,
                              void* d_out, int out_size, void* d_ws, size_t ws_size,
                              hipStream_t stream) {
  const float* img = (const float*)d_in[0];
  const float* txt = (const float*)d_in[1];
  const int D = 1024;
  const int B = in_sizes[0] / D;   // 8192
  const size_t ND = (size_t)B * D;

  // workspace layout: [row_max | col_max | pos | aHi | aLo | bHi | bLo]
  float* rm  = (float*)d_ws;
  float* cm  = rm + B;
  float* pos = cm + B;
  unsigned short* base = (unsigned short*)(pos + B);
  unsigned short* aHi = base;
  unsigned short* aLo = aHi + ND;
  unsigned short* bHi = aLo + ND;
  unsigned short* bLo = bHi + ND;

  init_ws<<<(B + 255) / 256, 256, 0, stream>>>(rm, cm, pos, B);

  int total4 = (int)(ND / 4);
  split_bf16<<<(total4 + 255) / 256, 256, 0, stream>>>(img, txt, aHi, aLo, bHi, bLo, total4);

  dim3 grid(B / TN, B / TM);
  tile_gemm_max<<<grid, 256, 0, stream>>>(aHi, aLo, bHi, bLo, rm, cm, pos, B, D);

  finalize_loss<<<1, 256, 0, stream>>>(rm, cm, pos, (float*)d_out, B);
}
